// GNNModel_59992103191101
// MI455X (gfx1250) — matrix operand order repaired and checked by reading, not compiled, not run on hardware
//
#include <hip/hip_runtime.h>

// ---------------------------------------------------------------------------
// Problem constants (match reference)
// ---------------------------------------------------------------------------
#define N_NODES   100000
#define N_EDGES   1600000
#define N_GRAPHS  64
#define F_IN      128
#define HDIM      64
#define H2DIM     128
#define IMG       50176
#define MLP0      1024
#define N_CLASSES 38

#define CDIV(a, b) (((a) + (b) - 1) / (b))

typedef __bf16 v16bf __attribute__((ext_vector_type(16)));
typedef float  v8f   __attribute__((ext_vector_type(8)));

__device__ __forceinline__ void atomAddF(float* p, float v) {
    unsafeAtomicAdd(p, v);   // -> global_atomic_add_f32 on gfx1250
}

// ---------------------------------------------------------------------------
// WMMA GEMM:  C[M,NREAL] (+)= A[M,K](f32) * B[K,NREAL](f32); bf16 inputs,
// f32 accumulate via v_wmma_f32_16x16x32_bf16.
//   MT    : m-tiles per wave (B-fragment reuse); M % (16*MT) == 0
//   NPAD  : N rounded up to multiple of 16 (compile time)
//   NREAL : actual N = B/C column count & row stride (compile time)
//   K     : reduction length = A row stride (compile time, % 32 == 0)
// gridDim.z == 1 : direct store, C = acc + bias[n] (bias may be null)
// gridDim.z  > 1 : split-K, atomicAdd into pre-initialized C
// Compile-time strides => all inner-loop loads/stores use immediate offsets.
// ---------------------------------------------------------------------------
template <int MT, int NPAD, int NREAL, int K>
__global__ void wmma_gemm(const float* __restrict__ A, const float* __restrict__ B,
                          const float* __restrict__ bias, float* __restrict__ C,
                          int M, int kChunk)
{
    constexpr int Ntiles = NPAD / 16;
    const int lane = threadIdx.x & 31;
    const int wid  = (int)((blockIdx.x * blockDim.x + threadIdx.x) >> 5);
    const int mGroups = M / (16 * MT);
    if (wid >= mGroups * Ntiles) return;

    const int mg = wid / Ntiles;
    const int nt = wid % Ntiles;
    const int lh  = lane >> 4;       // lane half (0/1)
    const int l16 = lane & 15;
    const int ncol = nt * 16 + l16;

    // Branchless column guard (only instantiated for the padded Wfc GEMM)
    bool nok = true;
    int  nc  = ncol;
    if constexpr (NREAL < NPAD) {
        nok = (ncol < NREAL);
        nc  = nok ? ncol : 0;
    }

    int k0 = (int)blockIdx.z * kChunk;
    int k1 = k0 + kChunk; if (k1 > K) k1 = K;

    v8f c[MT] = {};

    const float* bcol = B + nc;                     // column nc, row stride NREAL
    for (int k = k0; k < k1; k += 32) {
        // ---- B fragment: lane half lh holds rows K = k + lh*16 + j
        const float* bp = bcol + (size_t)(k + lh * 16) * NREAL;
        v16bf bf;
#pragma unroll
        for (int j = 0; j < 16; ++j) {
            float v = bp[j * NREAL];                // immediate offset j*NREAL*4
            if constexpr (NREAL < NPAD) v = nok ? v : 0.0f;
            bf[j] = (__bf16)v;
        }
        if (k + 32 < k1)
            __builtin_prefetch(bp + (size_t)32 * NREAL, 0, 1);

        // ---- A fragments + WMMA per m-tile (reuse B fragment MT times)
#pragma unroll
        for (int i = 0; i < MT; ++i) {
            const int mrow = (mg * MT + i) * 16 + l16;
            const float* arow = A + (size_t)mrow * K + k;
            const float4 a0 = *(const float4*)(arow + lh * 16);
            const float4 a1 = *(const float4*)(arow + lh * 16 + 4);
            const float4 a2 = *(const float4*)(arow + lh * 16 + 8);
            const float4 a3 = *(const float4*)(arow + lh * 16 + 12);
            v16bf af;
            af[0]=(__bf16)a0.x;  af[1]=(__bf16)a0.y;  af[2]=(__bf16)a0.z;  af[3]=(__bf16)a0.w;
            af[4]=(__bf16)a1.x;  af[5]=(__bf16)a1.y;  af[6]=(__bf16)a1.z;  af[7]=(__bf16)a1.w;
            af[8]=(__bf16)a2.x;  af[9]=(__bf16)a2.y;  af[10]=(__bf16)a2.z; af[11]=(__bf16)a2.w;
            af[12]=(__bf16)a3.x; af[13]=(__bf16)a3.y; af[14]=(__bf16)a3.z; af[15]=(__bf16)a3.w;
            c[i] = __builtin_amdgcn_wmma_f32_16x16x32_bf16(
                       false, af, false, bf, (short)0, c[i], false, false);
        }
    }

    if constexpr (NREAL < NPAD) { if (!nok) return; }
    const float bv = bias ? bias[ncol] : 0.0f;
#pragma unroll
    for (int i = 0; i < MT; ++i) {
#pragma unroll
        for (int r = 0; r < 8; ++r) {
            // C layout: accumulator element r -> M = r + lh*8 within the tile
            const int mrow = (mg * MT + i) * 16 + lh * 8 + r;
            float* dst = C + (size_t)mrow * NREAL + ncol;
            if (gridDim.z == 1) *dst = c[i][r] + bv;
            else                atomAddF(dst, c[i][r]);
        }
    }
}

// ---------------------------------------------------------------------------
// Small utility kernels
// ---------------------------------------------------------------------------
__global__ void fill_val(float* __restrict__ p, float v, int n) {
    int i = blockIdx.x * blockDim.x + threadIdx.x;
    if (i < n) p[i] = v;
}

__global__ void fill_rows(float* __restrict__ p, const float* __restrict__ bias,
                          int rows, int cols) {
    int i = blockIdx.x * blockDim.x + threadIdx.x;
    if (i < rows * cols) p[i] = bias[i % cols];
}

__global__ void edge_deg(float* __restrict__ deg, const int* __restrict__ dst, int E) {
    int e = blockIdx.x * blockDim.x + threadIdx.x;
    if (e < E) atomAddF(&deg[dst[e]], 1.0f);
}

__global__ void rsqrt_k(float* __restrict__ p, int n) {
    int i = blockIdx.x * blockDim.x + threadIdx.x;
    if (i < n) p[i] = rsqrtf(p[i]);   // deg >= 1 always (self loop)
}

// one wave per edge; lane strides features. agg[d] += h[s] * dinv[s]*dinv[d]
__global__ void scatter_edges(const float* __restrict__ h, const int* __restrict__ src,
                              const int* __restrict__ dst, const float* __restrict__ dinv,
                              float* __restrict__ agg, int E, int F)
{
    int wid  = (int)((blockIdx.x * blockDim.x + threadIdx.x) >> 5);
    int lane = threadIdx.x & 31;
    if (wid >= E) return;
    int s = src[wid], d = dst[wid];
    float nrm = dinv[s] * dinv[d];
    const float* hs = h + (size_t)s * F;
    float*       ad = agg + (size_t)d * F;
    for (int f = lane; f < F; f += 32)
        atomAddF(&ad[f], hs[f] * nrm);
}

// h = relu(agg + dinv[v]^2 * hlin + bias)  (self-loop term + bias), in place in agg
__global__ void finalize_gcn(const float* __restrict__ hlin, const float* __restrict__ dinv,
                             const float* __restrict__ bias, float* __restrict__ agg,
                             int Nn, int F)
{
    int i = blockIdx.x * blockDim.x + threadIdx.x;
    if (i >= Nn * F) return;
    int v = i / F, f = i % F;
    float di = dinv[v];
    float val = agg[i] + di * di * hlin[i] + bias[f];
    agg[i] = val > 0.0f ? val : 0.0f;
}

__global__ void pool_nodes(const float* __restrict__ h, const int* __restrict__ batch,
                           float* __restrict__ g, int Nn, int F)
{
    int i = blockIdx.x * blockDim.x + threadIdx.x;
    if (i >= Nn * F) return;
    int v = i / F, f = i % F;
    atomAddF(&g[(size_t)batch[v] * F + f], h[i]);
}

__global__ void concat_k(const float* __restrict__ a, const float* __restrict__ b,
                         float* __restrict__ o, int rows, int ca, int cb)
{
    int i = blockIdx.x * blockDim.x + threadIdx.x;
    int cols = ca + cb;
    if (i >= rows * cols) return;
    int r = i / cols, c = i % cols;
    o[i] = (c < ca) ? a[r * ca + c] : b[r * cb + (c - ca)];
}

// one wave per row; log_softmax over `cols`
__global__ void logsoftmax_k(const float* __restrict__ in, float* __restrict__ out,
                             int rows, int cols)
{
    int wid  = (int)((blockIdx.x * blockDim.x + threadIdx.x) >> 5);
    int lane = threadIdx.x & 31;
    if (wid >= rows) return;
    const float* row = in + (size_t)wid * cols;
    float m = -3.4e38f;
    for (int c = lane; c < cols; c += 32) m = fmaxf(m, row[c]);
    for (int off = 16; off > 0; off >>= 1) m = fmaxf(m, __shfl_xor(m, off, 32));
    float s = 0.0f;
    for (int c = lane; c < cols; c += 32) s += __expf(row[c] - m);
    for (int off = 16; off > 0; off >>= 1) s += __shfl_xor(s, off, 32);
    float ls = __logf(s);
    for (int c = lane; c < cols; c += 32)
        out[(size_t)wid * cols + c] = row[c] - m - ls;
}

// ---------------------------------------------------------------------------
// Orchestration
// ---------------------------------------------------------------------------
extern "C" void kernel_launch(void* const* d_in, const int* in_sizes, int n_in,
                              void* d_out, int out_size, void* d_ws, size_t ws_size,
                              hipStream_t stream)
{
    (void)in_sizes; (void)n_in; (void)out_size; (void)ws_size;

    const float* x    = (const float*)d_in[0];
    const int*   src  = (const int*)d_in[1];
    const int*   dst  = src + N_EDGES;
    const float* img  = (const float*)d_in[2];
    const int*   batch= (const int*)d_in[3];
    const float* W1   = (const float*)d_in[4];
    const float* b1   = (const float*)d_in[5];
    const float* W2   = (const float*)d_in[6];
    const float* b2   = (const float*)d_in[7];
    const float* Wm0  = (const float*)d_in[8];
    const float* bm0  = (const float*)d_in[9];
    const float* Wm1  = (const float*)d_in[10];
    const float* bm1  = (const float*)d_in[11];
    const float* Wmx  = (const float*)d_in[12];
    const float* bmx  = (const float*)d_in[13];
    const float* Wfc  = (const float*)d_in[14];
    const float* bfc  = (const float*)d_in[15];
    float* out = (float*)d_out;

    // ---- workspace layout (bytes, 16B aligned) --------------------------
    char* ws = (char*)d_ws;
    float* hlin  = (float*)(ws + 0);           // 51.2 MB: hlin1, then hlin2
    float* agg   = (float*)(ws + 51200000);    // 51.2 MB: agg1/h1, then agg2/h2
    float* dinv  = (float*)(ws + 102400000);   // 400 KB
    float* t0    = (float*)(ws + 102800000);   // 64*1024*4
    float* x0    = (float*)(ws + 103062144);   // 64*64*4
    float* g     = (float*)(ws + 103078528);   // 64*64*4
    float* g2    = (float*)(ws + 103094912);   // 64*64*4
    float* xt    = (float*)(ws + 103111296);   // 64*128*4
    float* logit = (float*)(ws + 103144064);   // 64*38*4

    const int B = 256;   // 8 waves per block

    // ---- degrees / normalization ---------------------------------------
    fill_val <<<CDIV(N_NODES, B), B, 0, stream>>>(dinv, 1.0f, N_NODES);
    edge_deg <<<CDIV(N_EDGES, B), B, 0, stream>>>(dinv, dst, N_EDGES);
    rsqrt_k  <<<CDIV(N_NODES, B), B, 0, stream>>>(dinv, N_NODES);

    // ---- GCN layer 1: hlin1 = x @ W1 ; aggregate ; relu(+b1) -----------
    // tiles = (100000/16)*(128/16) = 50000 waves
    wmma_gemm<1, H2DIM, H2DIM, F_IN><<<dim3(CDIV(50000, 8), 1, 1), B, 0, stream>>>(
        x, W1, nullptr, hlin, N_NODES, F_IN);
    fill_val <<<CDIV(N_NODES * H2DIM, B), B, 0, stream>>>(agg, 0.0f, N_NODES * H2DIM);
    scatter_edges <<<CDIV(N_EDGES, 8), B, 0, stream>>>(hlin, src, dst, dinv, agg,
                                                       N_EDGES, H2DIM);
    finalize_gcn <<<CDIV(N_NODES * H2DIM, B), B, 0, stream>>>(hlin, dinv, b1, agg,
                                                              N_NODES, H2DIM);  // agg = h1

    // ---- GCN layer 2: hlin2 = h1 @ W2 ; aggregate ; relu(+b2) ----------
    // tiles = 6250*4 = 25000 waves
    wmma_gemm<1, HDIM, HDIM, H2DIM><<<dim3(CDIV(25000, 8), 1, 1), B, 0, stream>>>(
        agg, W2, nullptr, hlin, N_NODES, H2DIM);
    fill_val <<<CDIV(N_NODES * HDIM, B), B, 0, stream>>>(agg, 0.0f, N_NODES * HDIM);
    scatter_edges <<<CDIV(N_EDGES, 8), B, 0, stream>>>(hlin, src, dst, dinv, agg,
                                                       N_EDGES, HDIM);
    finalize_gcn <<<CDIV(N_NODES * HDIM, B), B, 0, stream>>>(hlin, dinv, b2, agg,
                                                             N_NODES, HDIM);    // agg = h2

    // ---- pool + g @ Wmx + bmx ------------------------------------------
    fill_val  <<<CDIV(N_GRAPHS * HDIM, B), B, 0, stream>>>(g, 0.0f, N_GRAPHS * HDIM);
    pool_nodes<<<CDIV(N_NODES * HDIM, B), B, 0, stream>>>(agg, batch, g, N_NODES, HDIM);
    wmma_gemm<4, HDIM, HDIM, HDIM><<<dim3(1, 1, 1), B, 0, stream>>>(
        g, Wmx, bmx, g2, N_GRAPHS, HDIM);

    // ---- image branch: t0 = img @ Wm0 + bm0 ; x0 = t0 @ Wm1 + bm1 ------
    fill_rows <<<CDIV(N_GRAPHS * MLP0, B), B, 0, stream>>>(t0, bm0, N_GRAPHS, MLP0);
    // M=64 (1 group of MT=4), 64 n-tiles, 49 k-chunks of 1024 -> split-K atomics
    wmma_gemm<4, MLP0, MLP0, IMG><<<dim3(CDIV(64, 8), 1, 49), B, 0, stream>>>(
        img, Wm0, nullptr, t0, N_GRAPHS, 1024);
    fill_rows <<<CDIV(N_GRAPHS * HDIM, B), B, 0, stream>>>(x0, bm1, N_GRAPHS, HDIM);
    wmma_gemm<4, HDIM, HDIM, MLP0><<<dim3(1, 1, 4), B, 0, stream>>>(
        t0, Wm1, nullptr, x0, N_GRAPHS, 256);

    // ---- head: concat, fc, log_softmax ---------------------------------
    concat_k <<<CDIV(N_GRAPHS * (HDIM + HDIM), B), B, 0, stream>>>(
        x0, g2, xt, N_GRAPHS, HDIM, HDIM);
    wmma_gemm<4, 48, N_CLASSES, 2 * HDIM><<<dim3(1, 1, 1), B, 0, stream>>>(
        xt, Wfc, bfc, logit, N_GRAPHS, 2 * HDIM);
    logsoftmax_k <<<CDIV(N_GRAPHS * 32, B), B, 0, stream>>>(logit, out,
                                                            N_GRAPHS, N_CLASSES);
}
